// LMUCell_549755814158
// MI455X (gfx1250) — matrix operand order repaired and checked by reading, not compiled, not run on hardware
//
#include <hip/hip_runtime.h>
#include <hip/hip_bf16.h>
#include <math.h>

// ---------------------------------------------------------------------------
// LMU cell on MI455X (gfx1250, wave32, WMMA).
// Serial scan c<-c*A^T + Bu_t over 2048 steps is re-expressed with matrix
// powers (chunk L=64, 32 chunks) so everything becomes wide bf16 WMMA GEMMs:
//   S_j = sum_i Bu[64j+i] * R^(63-i)        (one GEMM, K = 64*512)
//   c_T = sum_j S_j * (R^64)^(31-j)         (one GEMM, K = 32*512)
// with power stacks A^0..A^63 and Q^0..Q^31 (Q=A^64) built by doubling.
// V3: register-double-buffered staging (next panel's global loads issue
// before the current panel's 32 WMMAs -> HBM latency hidden, single 68KB
// LDS buffer keeps 4 blocks/WGP), hoisted stage-invariant address math.
// Workspace use: ~262 MB.
// ---------------------------------------------------------------------------

typedef __bf16 bf16_t;
typedef __attribute__((ext_vector_type(16))) __bf16 v16bf;
typedef __attribute__((ext_vector_type(8)))  __bf16 v8bf;
typedef __attribute__((ext_vector_type(8)))  float  v8f;
typedef __attribute__((ext_vector_type(4)))  float  f32x4;

#define SEQ_T  2048
#define CHUNK  64
#define NCH    32
#define KT     128          // K panel staged in LDS per stage
#define LDSP   136          // LDS pitch in bf16 elems (272B: 16B-aligned, bank-offset 4)

struct GemmArgs {
  const void*   A;        // bf16 or f32 depending on AF32
  const bf16_t* Bt;       // [N x K] with per-512-K-block stride bt_kb
  const float*  bias;     // len N, or null
  const float*  accin;    // [M x N] f32, or null
  float*        out_f32;  // [M x N], or null
  bf16_t*       out_bf16; // [M x N], or null
  long lda;               // A row stride in elements (AMODE==0)
  long a_kb;              // A stride per 512-elem K block (may be negative)
  long ldb;               // Bt row stride in elements
  long bt_kb;             // Bt stride per 512-elem K block
  int M, N, K;            // K must be a multiple of 128
  int a_is_f32;
  int a_mode;             // 0: affine rows; 1: chunk-reversed Bu rows
  int act_tanh;
};

template <int AMODE>
__device__ __forceinline__ long a_rowbase(int row, long lda) {
  if (AMODE == 0) return (long)row * lda;
  // row = b*32 + j ; base at last timestep of chunk j, batch b
  const int b = row >> 5, j = row & 31;
  return ((long)b * SEQ_T + (long)j * CHUNK + (CHUNK - 1)) * 512;
}

// ---------------------------------------------------------------------------
// bf16 WMMA GEMM:  out[m,n] = act( sum_k A[m,k]*Bt[n,k] + bias + acc )
// 256 threads = 8 waves; block tile 128x128; wave tile 64x32; LDS-staged
// K=128 panels with register double-buffered global prefetch.
// ---------------------------------------------------------------------------
template <int AF32, int AMODE>
__launch_bounds__(256)
__global__ void gemm_wmma_bf16(GemmArgs g) {
  __shared__ bf16_t ldsA[128 * LDSP];
  __shared__ bf16_t ldsB[128 * LDSP];

  const int t    = threadIdx.x;
  const int lane = t & 31;
  const int wave = t >> 5;
  const int half = lane >> 4;   // 0: lanes 0-15, 1: lanes 16-31
  const int l16  = lane & 15;

  const int m_loc = (wave >> 2) * 64;   // wave origin inside block tile
  const int n_loc = (wave & 3) * 32;
  const int m_blk = blockIdx.y * 128;
  const int n_blk = blockIdx.x * 128;

  // chunk map: this thread stages chunks c = t + cc*256, c in [0,1024):
  //   panel row r = c>>3 (0..127), 16B chunk kc = (c&7)*8
  int  rA[4];        // LDS element offset of each staged chunk
  long rbA[4], rbB[4];  // stage-invariant global row bases
#pragma unroll
  for (int cc = 0; cc < 4; ++cc) {
    const int c = t + cc * 256;
    const int r = c >> 3;
    const int kc = (c & 7) * 8;
    rA[cc] = r * LDSP + kc;
    int row = m_blk + r;
    if (row >= g.M) row = g.M - 1;      // clamp (stores guarded)
    rbA[cc] = a_rowbase<AMODE>(row, g.lda);
    rbB[cc] = (long)(n_blk + r) * g.ldb;  // N is always a multiple of 128
  }

  v8f acc[4][2];
#pragma unroll
  for (int i = 0; i < 4; ++i)
#pragma unroll
    for (int j = 0; j < 2; ++j)
#pragma unroll
      for (int r = 0; r < 8; ++r) acc[i][j][r] = 0.0f;

  // ---- global fetch of one K-panel into registers
  v8bf sa[4], sb[4];
  auto fetch = [&](int kt0) {
#pragma unroll
    for (int cc = 0; cc < 4; ++cc) {
      const int c  = t + cc * 256;
      const int kk = kt0 + (c & 7) * 8;
      const long ga = rbA[cc] + (long)(kk >> 9) * g.a_kb + (kk & 511);
      if (AF32) {
        const float* Af = (const float*)g.A;
        f32x4 p0 = *(const f32x4*)(Af + ga);
        f32x4 p1 = *(const f32x4*)(Af + ga + 4);
        v8bf v;
#pragma unroll
        for (int e = 0; e < 4; ++e) {
          v[e]     = (bf16_t)p0[e];
          v[4 + e] = (bf16_t)p1[e];
        }
        sa[cc] = v;
      } else {
        sa[cc] = *(const v8bf*)((const bf16_t*)g.A + ga);
      }
      const long gb = rbB[cc] + (long)(kk >> 9) * g.bt_kb + (kk & 511);
      sb[cc] = *(const v8bf*)(g.Bt + gb);
    }
  };

  fetch(0);

  for (int kt0 = 0; kt0 < g.K; kt0 += KT) {
    __syncthreads();   // all waves done reading the previous panel
#pragma unroll
    for (int cc = 0; cc < 4; ++cc) {
      *(v8bf*)&ldsA[rA[cc]] = sa[cc];
      *(v8bf*)&ldsB[rA[cc]] = sb[cc];
    }
    __syncthreads();

    // prefetch next panel into registers; loads fly over the 32 WMMAs below
    if (kt0 + KT < g.K) fetch(kt0 + KT);

    // ---- 4 x (K=32) WMMA substeps from LDS
#pragma unroll
    for (int s = 0; s < 4; ++s) {
      const int kk0 = s * 32;
      // A fragments (16x32): lanes 0-15 K [kk0..+7],[+16..+23]; lanes 16-31 +8
      v16bf af[4];
#pragma unroll
      for (int i = 0; i < 4; ++i) {
        const int ro  = (m_loc + i * 16 + l16) * LDSP;
        const int klo = kk0 + half * 8;
        v8bf lo = *(const v8bf*)&ldsA[ro + klo];
        v8bf hi = *(const v8bf*)&ldsA[ro + klo + 16];
        v16bf a;
#pragma unroll
        for (int e = 0; e < 8; ++e) { a[e] = lo[e]; a[8 + e] = hi[e]; }
        af[i] = a;
      }
      // B fragments (same K order as A): lanes 0-15 K [kk0..+7],[+16..+23]; lanes 16-31 +8
      v16bf bfr[2];
#pragma unroll
      for (int j = 0; j < 2; ++j) {
        const int ro = (n_loc + j * 16 + l16) * LDSP;
        const int ks = kk0 + half * 8;
        v8bf b0 = *(const v8bf*)&ldsB[ro + ks];
        v8bf b1 = *(const v8bf*)&ldsB[ro + ks + 16];
        v16bf b;
#pragma unroll
        for (int e = 0; e < 8; ++e) { b[e] = b0[e]; b[8 + e] = b1[e]; }
        bfr[j] = b;
      }
#pragma unroll
      for (int i = 0; i < 4; ++i)
#pragma unroll
        for (int j = 0; j < 2; ++j)
          acc[i][j] = __builtin_amdgcn_wmma_f32_16x16x32_bf16(
              false, af[i], false, bfr[j], (short)0, acc[i][j], false, false);
    }
  }

  // ---- Epilogue. C/D layout: VGPR r -> M = r + (half?8:0), N = l16.
#pragma unroll
  for (int i = 0; i < 4; ++i) {
#pragma unroll
    for (int j = 0; j < 2; ++j) {
      const int n = n_blk + n_loc + j * 16 + l16;
#pragma unroll
      for (int r = 0; r < 8; ++r) {
        const int m = m_blk + m_loc + i * 16 + r + half * 8;
        if (m >= g.M) continue;
        float v = acc[i][j][r];
        if (g.bias)  v += g.bias[n];
        if (g.accin) v += g.accin[(long)m * g.N + n];
        if (g.act_tanh) v = tanhf(v);
        const long o = (long)m * g.N + n;
        if (g.out_f32)  g.out_f32[o]  = v;
        if (g.out_bf16) g.out_bf16[o] = (bf16_t)v;
      }
    }
  }
}

// ---------------------------------------------------------------------------
// Small helper kernels
// ---------------------------------------------------------------------------
__global__ void cast_f32_bf16(const float* s, bf16_t* d, long n) {
  long i = (long)blockIdx.x * blockDim.x + threadIdx.x;
  const long stride = (long)gridDim.x * blockDim.x;
  for (; i < n; i += stride) d[i] = (bf16_t)s[i];
}

__global__ void transpose512_bf16(const bf16_t* s, bf16_t* d) {
  const int idx = blockIdx.x * blockDim.x + threadIdx.x;  // 0..262143
  const int i = idx >> 9, j = idx & 511;
  d[idx] = s[(long)j * 512 + i];
}

__global__ void transpose512_cast(const float* s, bf16_t* d) {
  const int idx = blockIdx.x * blockDim.x + threadIdx.x;
  const int i = idx >> 9, j = idx & 511;
  d[idx] = (bf16_t)s[(long)j * 512 + i];
}

__global__ void identity512_bf16(bf16_t* d) {
  const int idx = blockIdx.x * blockDim.x + threadIdx.x;
  const int i = idx >> 9, j = idx & 511;
  d[idx] = (i == j) ? (bf16_t)1.0f : (bf16_t)0.0f;
}

// ---------------------------------------------------------------------------
static void launch_gemm(const GemmArgs& g, hipStream_t stream) {
  const dim3 grid((unsigned)(g.N / 128), (unsigned)((g.M + 127) / 128));
  const dim3 blk(256);
  if (g.a_is_f32)         gemm_wmma_bf16<1, 0><<<grid, blk, 0, stream>>>(g);
  else if (g.a_mode == 1) gemm_wmma_bf16<0, 1><<<grid, blk, 0, stream>>>(g);
  else                    gemm_wmma_bf16<0, 0><<<grid, blk, 0, stream>>>(g);
}

// ---------------------------------------------------------------------------
extern "C" void kernel_launch(void* const* d_in, const int* in_sizes, int n_in,
                              void* d_out, int out_size, void* d_ws, size_t ws_size,
                              hipStream_t stream) {
  (void)in_sizes; (void)n_in; (void)out_size; (void)ws_size;
  const float* x   = (const float*)d_in[0];  // [32,2048,1024]
  const float* A   = (const float*)d_in[1];  // [512,512]
  const float* B   = (const float*)d_in[2];  // [512,1024]
  const float* C   = (const float*)d_in[3];  // [1024,512]
  const float* D   = (const float*)d_in[4];  // [1024,1024]
  const float* Win = (const float*)d_in[5];  // [1024,1024]
  const float* bin = (const float*)d_in[6];  // [1024]
  float* out = (float*)d_out;                // [32,1024]

  const long BATCH = 32, SEQ = 2048, INP = 1024, HID = 1024, MEM = 512;
  const long BS   = BATCH * SEQ;     // 65536
  const long MBLK = MEM * MEM;       // 262144 elems per 512x512 block

  // ---- workspace carve (~262 MB total)
  char* p = (char*)d_ws;
  auto carve = [&](long bytes) -> void* {
    void* r = (void*)p; p += (bytes + 255) & ~255L; return r;
  };
  bf16_t* u_b   = (bf16_t*)carve(BS * HID * 2);     // 128 MB
  bf16_t* Bu_b  = (bf16_t*)carve(BS * MEM * 2);     // 64 MB
  bf16_t* Win_b = (bf16_t*)carve(HID * INP * 2);
  bf16_t* B_b   = (bf16_t*)carve(MEM * INP * 2);
  bf16_t* C_b   = (bf16_t*)carve(HID * MEM * 2);
  bf16_t* D_b   = (bf16_t*)carve(HID * INP * 2);
  bf16_t* R_b   = (bf16_t*)carve(MBLK * 2);         // A^T
  bf16_t* APow  = (bf16_t*)carve(64 * MBLK * 2);    // A^0..A^63, 32 MB
  bf16_t* QPow  = (bf16_t*)carve(32 * MBLK * 2);    // Q^0..Q^31, 16 MB
  bf16_t* T_b   = (bf16_t*)carve(MBLK * 2);         // transpose scratch
  bf16_t* QT_b  = (bf16_t*)carve(MBLK * 2);         // Q^T
  bf16_t* S_b   = (bf16_t*)carve(BATCH * NCH * MEM * 2);
  bf16_t* cT_b  = (bf16_t*)carve(BATCH * MEM * 2);
  float*  hacc  = (float*)carve(BATCH * HID * 4);

  const dim3 tb(256);

  // ---- weight / operand conversions
  cast_f32_bf16<<<1024, tb, 0, stream>>>(Win, Win_b, HID * INP);
  cast_f32_bf16<<<512,  tb, 0, stream>>>(B,   B_b,   MEM * INP);
  cast_f32_bf16<<<512,  tb, 0, stream>>>(C,   C_b,   HID * MEM);
  cast_f32_bf16<<<1024, tb, 0, stream>>>(D,   D_b,   HID * INP);
  cast_f32_bf16<<<256,  tb, 0, stream>>>(A,   APow + MBLK, MEM * MEM);  // A^1
  transpose512_cast<<<1024, tb, 0, stream>>>(A, R_b);                   // A^T
  identity512_bf16<<<1024, tb, 0, stream>>>(APow);                      // A^0
  identity512_bf16<<<1024, tb, 0, stream>>>(QPow);                      // Q^0

  // ---- u = tanh(x @ Win^T + b)   (reads f32 x, converts while staging)
  { GemmArgs g = {};
    g.A = x; g.a_is_f32 = 1; g.lda = INP; g.a_kb = 512;
    g.Bt = Win_b; g.ldb = INP; g.bt_kb = 512;
    g.bias = bin; g.act_tanh = 1; g.out_bf16 = u_b;
    g.M = (int)BS; g.N = (int)HID; g.K = (int)INP;
    launch_gemm(g, stream); }

  // ---- Bu = u @ B^T
  { GemmArgs g = {};
    g.A = u_b; g.lda = HID; g.a_kb = 512;
    g.Bt = B_b; g.ldb = INP; g.bt_kb = 512;
    g.out_bf16 = Bu_b;
    g.M = (int)BS; g.N = (int)MEM; g.K = (int)INP;
    launch_gemm(g, stream); }

  // ---- power table A^0..A^63 by doubling (5 rounds)
  for (int cnt = 2; cnt < 64; cnt *= 2) {
    GemmArgs g = {};
    // A^cnt = A^(cnt-1) @ A   (Bt = A^T)
    g.A = APow + (long)(cnt - 1) * MBLK; g.lda = 512; g.a_kb = 512;
    g.Bt = R_b; g.ldb = 512; g.bt_kb = 512;
    g.M = 512; g.N = 512; g.K = 512;
    g.out_bf16 = APow + (long)cnt * MBLK;
    launch_gemm(g, stream);
    transpose512_bf16<<<1024, tb, 0, stream>>>(APow + (long)cnt * MBLK, T_b);
    // A^(cnt+k) = A^k @ A^cnt  for k=1..cnt-1  (stacked rows = one GEMM)
    g.A = APow + MBLK; g.M = (cnt - 1) * 512; g.Bt = T_b;
    g.out_bf16 = APow + (long)(cnt + 1) * MBLK;
    launch_gemm(g, stream);
  }

  // ---- Q = A^64 = A^63 @ A ; then Q^0..Q^31 by doubling (4 rounds)
  { GemmArgs g = {};
    g.A = APow + 63L * MBLK; g.lda = 512; g.a_kb = 512;
    g.Bt = R_b; g.ldb = 512; g.bt_kb = 512;
    g.M = 512; g.N = 512; g.K = 512;
    g.out_bf16 = QPow + MBLK;
    launch_gemm(g, stream); }
  transpose512_bf16<<<1024, tb, 0, stream>>>(QPow + MBLK, QT_b);
  for (int cnt = 2; cnt < 32; cnt *= 2) {
    GemmArgs g = {};
    g.A = QPow + (long)(cnt - 1) * MBLK; g.lda = 512; g.a_kb = 512;
    g.Bt = QT_b; g.ldb = 512; g.bt_kb = 512;
    g.M = 512; g.N = 512; g.K = 512;
    g.out_bf16 = QPow + (long)cnt * MBLK;
    launch_gemm(g, stream);
    transpose512_bf16<<<1024, tb, 0, stream>>>(QPow + (long)cnt * MBLK, T_b);
    g.A = QPow + MBLK; g.M = (cnt - 1) * 512; g.Bt = T_b;
    g.out_bf16 = QPow + (long)(cnt + 1) * MBLK;
    launch_gemm(g, stream);
  }

  // ---- S[b,j] = sum_i Bu[b,64j+i] @ R^(63-i)
  // A-operand: chunk-reversed Bu rows; Bt block i' = A^(i') (ascending stack).
  { GemmArgs g = {};
    g.A = Bu_b; g.a_mode = 1; g.a_kb = -512;
    g.Bt = APow; g.ldb = 512; g.bt_kb = MBLK;
    g.M = (int)(BATCH * NCH); g.N = (int)MEM; g.K = CHUNK * (int)MEM;
    g.out_bf16 = S_b;
    launch_gemm(g, stream); }

  // ---- c_T[b] = sum_j S[b,j] @ (R^64)^(31-j)
  { GemmArgs g = {};
    g.A = S_b + (NCH - 1) * MEM; g.lda = NCH * MEM; g.a_kb = -512;
    g.Bt = QPow; g.ldb = 512; g.bt_kb = MBLK;
    g.M = (int)BATCH; g.N = (int)MEM; g.K = NCH * (int)MEM;
    g.out_bf16 = cT_b;
    launch_gemm(g, stream); }

  // ---- h = tanh(cT @ C^T + u_last @ D^T)
  { GemmArgs g = {};
    g.A = cT_b; g.lda = MEM; g.a_kb = 512;
    g.Bt = C_b; g.ldb = MEM; g.bt_kb = 512;
    g.M = (int)BATCH; g.N = (int)HID; g.K = (int)MEM;
    g.out_f32 = hacc;
    launch_gemm(g, stream); }
  { GemmArgs g = {};
    g.A = u_b + (SEQ - 1) * INP; g.lda = SEQ * INP; g.a_kb = 512;  // u_last rows
    g.Bt = D_b; g.ldb = INP; g.bt_kb = 512;
    g.M = (int)BATCH; g.N = (int)HID; g.K = (int)INP;
    g.accin = hacc; g.act_tanh = 1; g.out_f32 = out;
    launch_gemm(g, stream); }
}
